// CustomTransformerEncoderModel_62723702391523
// MI455X (gfx1250) — matrix operand order repaired and checked by reading, not compiled, not run on hardware
//
#include <hip/hip_runtime.h>
#include <hip/hip_bf16.h>
#include <math.h>

#define B_  2
#define S_  1024
#define V_  32000
#define D_  768
#define F_  3072
#define H_  12
#define L_  6
#define DH_ 64

typedef __attribute__((ext_vector_type(16))) __bf16 v16bf;
typedef __attribute__((ext_vector_type(8)))  float  v8f;

__device__ __forceinline__ float fcomp(const float4& f, int j) {
  return j == 0 ? f.x : (j == 1 ? f.y : (j == 2 ? f.z : f.w));
}

// A-fragment (contiguous k order chosen here, not the ISA table order): one contiguous 16-elem run at
// k = half*16 .. +15.  p points at row start (+kk).
__device__ __forceinline__ v16bf ld_a_frag(const __bf16* p, int half) {
  v16bf r;
  __builtin_memcpy(&r, p + half * 16, 32);
  return r;
}
// B-fragment (same contiguous k order as A, not the ISA table order): one contiguous 16-elem run.
__device__ __forceinline__ v16bf ld_b_frag(const __bf16* p) {
  v16bf r;
  __builtin_memcpy(&r, p, 32);
  return r;
}

// ---------------------------------------------------------------------------
// Generic GEMM: C[M,N] = act(A[M,K] * W[K,N] + bias[N])
// block tile 64(M) x 128(N), BK=32, 256 threads = 8 waves.
// Double-buffered LDS; next tile staged in registers while WMMAs run.
// W tile stored TRANSPOSED in LDS (K-contiguous rows) so B-fragments are
// contiguous b128 loads and the transpose store is 4x ds_store_b64.
// ---------------------------------------------------------------------------
__global__ void __launch_bounds__(256)
gemm_bf16_wmma(const float* __restrict__ A, const float* __restrict__ W,
               const float* __restrict__ bias, float* __restrict__ C,
               int M, int K, int N, int act) {
  __shared__ alignas(16) __bf16 As[2][64][40];    // [m][k], 80B row stride
  __shared__ alignas(16) __bf16 BsT[2][128][40];  // [n][k], 80B row stride
  const int tid  = threadIdx.x;
  const int lane = tid & 31, wave = tid >> 5;
  const int half = lane >> 4, lx = lane & 15;
  const int mt = wave & 3, ng = wave >> 2;
  const int bm = blockIdx.y * 64, bn = blockIdx.x * 128;
  // per-thread tile-load coordinates (fixed across K-steps)
  const int rA = tid >> 3, c4A = tid & 7;        // A: rows rA, rA+32; col grp c4A
  const int rB = (tid >> 5) * 4, c4B = tid & 31; // W: rows rB..rB+3; col grp c4B

  v8f acc[4];
#pragma unroll
  for (int t = 0; t < 4; ++t)
#pragma unroll
    for (int r = 0; r < 8; ++r) acc[t][r] = 0.0f;

  // ---- prologue: stage tile k=0 into buffer 0
  {
    float4 a0 = *(const float4*)(A + (size_t)(bm + rA) * K + c4A * 4);
    float4 a1 = *(const float4*)(A + (size_t)(bm + rA + 32) * K + c4A * 4);
    __bf16 s0[4] = {(__bf16)a0.x, (__bf16)a0.y, (__bf16)a0.z, (__bf16)a0.w};
    __bf16 s1[4] = {(__bf16)a1.x, (__bf16)a1.y, (__bf16)a1.z, (__bf16)a1.w};
    __builtin_memcpy(&As[0][rA][c4A * 4], s0, 8);
    __builtin_memcpy(&As[0][rA + 32][c4A * 4], s1, 8);
    float4 rw[4];
#pragma unroll
    for (int i = 0; i < 4; ++i)
      rw[i] = *(const float4*)(W + (size_t)(rB + i) * N + bn + c4B * 4);
#pragma unroll
    for (int j = 0; j < 4; ++j) {   // K-contiguous pack per output column
      __bf16 s[4] = {(__bf16)fcomp(rw[0], j), (__bf16)fcomp(rw[1], j),
                     (__bf16)fcomp(rw[2], j), (__bf16)fcomp(rw[3], j)};
      __builtin_memcpy(&BsT[0][c4B * 4 + j][rB], s, 8);
    }
  }
  __syncthreads();

  int buf = 0;
  for (int k0 = 0; k0 < K; k0 += 32, buf ^= 1) {
    // clamp: last iteration redundantly re-stages its own tile (no divergence)
    const int kn = (k0 + 32 < K) ? (k0 + 32) : k0;
    // ---- issue next tile's global loads (overlap with WMMA below)
    float4 ra0 = *(const float4*)(A + (size_t)(bm + rA) * K + kn + c4A * 4);
    float4 ra1 = *(const float4*)(A + (size_t)(bm + rA + 32) * K + kn + c4A * 4);
    float4 rb[4];
#pragma unroll
    for (int i = 0; i < 4; ++i)
      rb[i] = *(const float4*)(W + (size_t)(kn + rB + i) * N + bn + c4B * 4);
    __builtin_prefetch(A + (size_t)(bm + rA) * K + k0 + 64 + c4A * 4);
    __builtin_prefetch(W + (size_t)(k0 + 64 + rB) * N + bn + c4B * 4);

    // ---- compute on current buffer
    v16bf a = ld_a_frag(&As[buf][mt * 16 + lx][0], half);
#pragma unroll
    for (int t = 0; t < 4; ++t) {
      int nt = ng * 4 + t;
      v16bf b = ld_b_frag(&BsT[buf][nt * 16 + lx][half * 16]);
      acc[t] = __builtin_amdgcn_wmma_f32_16x16x32_bf16(false, a, false, b,
                                                       (short)0, acc[t], false, false);
    }

    // ---- convert + store staged tile into the other buffer
    {
      __bf16 s0[4] = {(__bf16)ra0.x, (__bf16)ra0.y, (__bf16)ra0.z, (__bf16)ra0.w};
      __bf16 s1[4] = {(__bf16)ra1.x, (__bf16)ra1.y, (__bf16)ra1.z, (__bf16)ra1.w};
      __builtin_memcpy(&As[buf ^ 1][rA][c4A * 4], s0, 8);
      __builtin_memcpy(&As[buf ^ 1][rA + 32][c4A * 4], s1, 8);
#pragma unroll
      for (int j = 0; j < 4; ++j) {
        __bf16 s[4] = {(__bf16)fcomp(rb[0], j), (__bf16)fcomp(rb[1], j),
                       (__bf16)fcomp(rb[2], j), (__bf16)fcomp(rb[3], j)};
        __builtin_memcpy(&BsT[buf ^ 1][c4B * 4 + j][rB], s, 8);
      }
    }
    __syncthreads();
  }

#pragma unroll
  for (int t = 0; t < 4; ++t) {
    int nt = ng * 4 + t;
#pragma unroll
    for (int r = 0; r < 8; ++r) {
      int row = bm + mt * 16 + r + 8 * half;   // C layout: lanes 16-31 hold M+8
      int col = bn + nt * 16 + lx;
      float v = acc[t][r] + bias[col];
      if (act == 1) v = 0.5f * v * (1.0f + erff(v * 0.70710678118654752f));
      C[(size_t)row * N + col] = v;
    }
  }
}

// ---------------------------------------------------------------------------
// Token + positional embedding (float4)
// ---------------------------------------------------------------------------
__global__ void __launch_bounds__(256)
embed_kernel(const int* __restrict__ x, const float* __restrict__ tok,
             const float* __restrict__ pos, float* __restrict__ h) {
  int row = blockIdx.x;                 // 0 .. B*S-1
  int s = row % S_;
  int t = x[row];
  int c4 = threadIdx.x;
  if (c4 < D_ / 4) {
    float4 a = ((const float4*)(tok + (size_t)t * D_))[c4];
    float4 b = ((const float4*)(pos + (size_t)s * D_))[c4];
    float4 o;
    o.x = a.x + b.x; o.y = a.y + b.y; o.z = a.z + b.z; o.w = a.w + b.w;
    ((float4*)(h + (size_t)row * D_))[c4] = o;
  }
}

// ---------------------------------------------------------------------------
// out = LayerNorm(a + b) * g + be   (one row per block; 192 float4 lanes)
// ---------------------------------------------------------------------------
__global__ void __launch_bounds__(256)
add_layernorm_kernel(const float* __restrict__ a, const float* __restrict__ b,
                     const float* __restrict__ g, const float* __restrict__ be,
                     float* __restrict__ out) {
  __shared__ float red[256];
  int row = blockIdx.x, tid = threadIdx.x;
  const bool active = tid < D_ / 4;
  float4 v = {0.f, 0.f, 0.f, 0.f};
  if (active) {
    float4 x = ((const float4*)(a + (size_t)row * D_))[tid];
    float4 y = ((const float4*)(b + (size_t)row * D_))[tid];
    v.x = x.x + y.x; v.y = x.y + y.y; v.z = x.z + y.z; v.w = x.w + y.w;
  }
  red[tid] = v.x + v.y + v.z + v.w;
  __syncthreads();
  for (int o = 128; o > 0; o >>= 1) { if (tid < o) red[tid] += red[tid + o]; __syncthreads(); }
  float mean = red[0] * (1.0f / D_);
  __syncthreads();
  float s2 = 0.0f;
  if (active) {
    float dx = v.x - mean, dy = v.y - mean, dz = v.z - mean, dw = v.w - mean;
    s2 = dx * dx + dy * dy + dz * dz + dw * dw;
  }
  red[tid] = s2;
  __syncthreads();
  for (int o = 128; o > 0; o >>= 1) { if (tid < o) red[tid] += red[tid + o]; __syncthreads(); }
  float rstd = rsqrtf(red[0] * (1.0f / D_) + 1e-5f);
  if (active) {
    float4 gg = ((const float4*)g)[tid];
    float4 bb = ((const float4*)be)[tid];
    float4 o;
    o.x = (v.x - mean) * rstd * gg.x + bb.x;
    o.y = (v.y - mean) * rstd * gg.y + bb.y;
    o.z = (v.z - mean) * rstd * gg.z + bb.z;
    o.w = (v.w - mean) * rstd * gg.w + bb.w;
    ((float4*)(out + (size_t)row * D_))[tid] = o;
  }
}

// ---------------------------------------------------------------------------
// Flash attention, one (b,h,64-query-block) per workgroup, all tiles in LDS.
// V stored transposed (K-contiguous packs) so the P*V B-fragment is a
// contiguous LDS run and the transpose store is 4x ds_store_b64.
// ---------------------------------------------------------------------------
__global__ void __launch_bounds__(256)
flash_attn_kernel(const float* __restrict__ q, const float* __restrict__ k,
                  const float* __restrict__ v, float* __restrict__ ctx) {
  __shared__ alignas(16) __bf16 Qs[64][72];   // [q][dh]
  __shared__ alignas(16) __bf16 Ks[64][72];   // [key][dh]; reused as P[q][key]
  __shared__ alignas(16) __bf16 Vt[64][72];   // [dh][key] (transposed)
  __shared__ float Ss[64][64];
  __shared__ float Oa[64][64];
  __shared__ float mrow[64], lrow[64], corr_s[64];
  __shared__ float red4[64][4];

  const int tid = threadIdx.x, lane = tid & 31, wave = tid >> 5;
  const int half = lane >> 4, lx = lane & 15;
  const int mt = wave & 3, ng = wave >> 2;
  const int row4 = tid >> 2, qt = tid & 3;           // softmax: 4 threads per row
  const int rq = (tid >> 4) * 4, c4 = tid & 15;      // tile loads: 4 rows, col grp
  const int bb = blockIdx.y / H_, hh = blockIdx.y % H_;
  const int q0 = blockIdx.x * 64;
  const float scale = 0.03608439182435161f; // 1/sqrt(768)
  const size_t base = (size_t)(bb * S_) * D_ + (size_t)hh * DH_;

#pragma unroll
  for (int i = 0; i < 4; ++i) {
    int r = rq + i;
    float4 f = *(const float4*)(q + base + (size_t)(q0 + r) * D_ + c4 * 4);
    __bf16 s4[4] = {(__bf16)(f.x * scale), (__bf16)(f.y * scale),
                    (__bf16)(f.z * scale), (__bf16)(f.w * scale)};
    __builtin_memcpy(&Qs[r][c4 * 4], s4, 8);
    float4 z = {0.f, 0.f, 0.f, 0.f};
    ((float4*)&Oa[r][0])[c4] = z;
  }
  if (tid < 64) { mrow[tid] = -1e30f; lrow[tid] = 0.0f; }
  __syncthreads();

  for (int kb = 0; kb < S_; kb += 64) {
    float4 gv[4];
#pragma unroll
    for (int i = 0; i < 4; ++i) {
      int r = rq + i;
      float4 f = *(const float4*)(k + base + (size_t)(kb + r) * D_ + c4 * 4);
      __bf16 s4[4] = {(__bf16)f.x, (__bf16)f.y, (__bf16)f.z, (__bf16)f.w};
      __builtin_memcpy(&Ks[r][c4 * 4], s4, 8);
      gv[i] = *(const float4*)(v + base + (size_t)(kb + r) * D_ + c4 * 4);
    }
#pragma unroll
    for (int j = 0; j < 4; ++j) {     // V transpose: K-contiguous b64 packs
      __bf16 s[4] = {(__bf16)fcomp(gv[0], j), (__bf16)fcomp(gv[1], j),
                     (__bf16)fcomp(gv[2], j), (__bf16)fcomp(gv[3], j)};
      __builtin_memcpy(&Vt[c4 * 4 + j][rq], s, 8);
    }
    __syncthreads();

    // S = Q * K^T : wave handles tiles (mt, nt = ng*2 + t)
#pragma unroll
    for (int t = 0; t < 2; ++t) {
      int nt = ng * 2 + t;
      v8f acc;
#pragma unroll
      for (int r = 0; r < 8; ++r) acc[r] = 0.0f;
#pragma unroll
      for (int kk = 0; kk < 64; kk += 32) {
        v16bf a = ld_a_frag(&Qs[mt * 16 + lx][kk], half);
        v16bf b = ld_b_frag(&Ks[nt * 16 + lx][kk + half * 16]); // B[k][n]=K[n][k]
        acc = __builtin_amdgcn_wmma_f32_16x16x32_bf16(false, a, false, b,
                                                      (short)0, acc, false, false);
      }
#pragma unroll
      for (int r = 0; r < 8; ++r)
        Ss[mt * 16 + r + 8 * half][nt * 16 + lx] = acc[r];
    }
    __syncthreads();

    // online softmax, 4 threads per row, 16 columns each
    {
      float pm = -1e30f;
      const float4* srow = (const float4*)&Ss[row4][qt * 16];
#pragma unroll
      for (int i = 0; i < 4; ++i) {
        float4 f = srow[i];
        pm = fmaxf(pm, fmaxf(fmaxf(f.x, f.y), fmaxf(f.z, f.w)));
      }
      red4[row4][qt] = pm;
      __syncthreads();
      if (qt == 0) {
        float m_old = mrow[row4];
        float mx = fmaxf(fmaxf(red4[row4][0], red4[row4][1]),
                         fmaxf(red4[row4][2], red4[row4][3]));
        mx = fmaxf(mx, m_old);
        corr_s[row4] = __expf(m_old - mx);
        mrow[row4] = mx;
      }
      __syncthreads();
      float mx = mrow[row4], corr = corr_s[row4];
      float psum = 0.0f;
#pragma unroll
      for (int c = 0; c < 16; ++c) {
        int cc = qt * 16 + c;
        float p = __expf(Ss[row4][cc] - mx);
        psum += p;
        Ks[row4][cc] = (__bf16)p;        // P overwrites K tile
        Oa[row4][cc] *= corr;
      }
      red4[row4][qt] = psum;
      __syncthreads();
      if (qt == 0)
        lrow[row4] = lrow[row4] * corr +
                     red4[row4][0] + red4[row4][1] + red4[row4][2] + red4[row4][3];
      __syncthreads();
    }

    // O += P * V
#pragma unroll
    for (int t = 0; t < 2; ++t) {
      int nt = ng * 2 + t;
      v8f acc;
#pragma unroll
      for (int r = 0; r < 8; ++r) acc[r] = 0.0f;
#pragma unroll
      for (int kk = 0; kk < 64; kk += 32) {
        v16bf a = ld_a_frag(&Ks[mt * 16 + lx][kk], half);       // P[q][key]
        v16bf b = ld_b_frag(&Vt[nt * 16 + lx][kk + half * 16]); // B[k][n]=Vt[n][k]
        acc = __builtin_amdgcn_wmma_f32_16x16x32_bf16(false, a, false, b,
                                                      (short)0, acc, false, false);
      }
#pragma unroll
      for (int r = 0; r < 8; ++r)
        Oa[mt * 16 + r + 8 * half][nt * 16 + lx] += acc[r];
    }
    __syncthreads();
  }

#pragma unroll
  for (int i = 0; i < 4; ++i) {
    int r = rq + i;
    float inv = 1.0f / lrow[r];
    float4 f = ((const float4*)&Oa[r][0])[c4];
    f.x *= inv; f.y *= inv; f.z *= inv; f.w *= inv;
    *(float4*)(ctx + base + (size_t)(q0 + r) * D_ + c4 * 4) = f;
  }
}

// ---------------------------------------------------------------------------
// In-place log_softmax over V=32000, one row per block (float4)
// ---------------------------------------------------------------------------
__global__ void __launch_bounds__(256)
log_softmax_kernel(float* __restrict__ logits) {
  __shared__ float red[256];
  const int NV4 = V_ / 4;                 // 8000
  int row = blockIdx.x, tid = threadIdx.x;
  float4* p4 = (float4*)(logits + (size_t)row * V_);
  float mx = -1e30f;
  for (int i = tid; i < NV4; i += 256) {
    float4 f = p4[i];
    mx = fmaxf(mx, fmaxf(fmaxf(f.x, f.y), fmaxf(f.z, f.w)));
  }
  red[tid] = mx; __syncthreads();
  for (int o = 128; o > 0; o >>= 1) { if (tid < o) red[tid] = fmaxf(red[tid], red[tid + o]); __syncthreads(); }
  mx = red[0]; __syncthreads();
  float s = 0.0f;
  for (int i = tid; i < NV4; i += 256) {
    float4 f = p4[i];
    s += __expf(f.x - mx) + __expf(f.y - mx) + __expf(f.z - mx) + __expf(f.w - mx);
  }
  red[tid] = s; __syncthreads();
  for (int o = 128; o > 0; o >>= 1) { if (tid < o) red[tid] += red[tid + o]; __syncthreads(); }
  float lg = mx + logf(red[0]);
  for (int i = tid; i < NV4; i += 256) {
    float4 f = p4[i];
    f.x -= lg; f.y -= lg; f.z -= lg; f.w -= lg;
    p4[i] = f;
  }
}

__global__ void __launch_bounds__(256)
copy4_kernel(const float* __restrict__ src, float* __restrict__ dst, int n4) {
  int i = blockIdx.x * 256 + threadIdx.x;
  if (i < n4) ((float4*)dst)[i] = ((const float4*)src)[i];
}

// ---------------------------------------------------------------------------
extern "C" void kernel_launch(void* const* d_in, const int* in_sizes, int n_in,
                              void* d_out, int out_size, void* d_ws, size_t ws_size,
                              hipStream_t stream) {
  (void)in_sizes; (void)n_in; (void)out_size; (void)ws_size;
  const int*   x    = (const int*)d_in[0];
  const float* tok  = (const float*)d_in[1];
  const float* pos  = (const float*)d_in[2];
  const float* Wq   = (const float*)d_in[3];
  const float* bq   = (const float*)d_in[4];
  const float* Wk   = (const float*)d_in[5];
  const float* bk   = (const float*)d_in[6];
  const float* Wv   = (const float*)d_in[7];
  const float* bv   = (const float*)d_in[8];
  const float* Wo   = (const float*)d_in[9];
  const float* bo   = (const float*)d_in[10];
  const float* ln1g = (const float*)d_in[11];
  const float* ln1b = (const float*)d_in[12];
  const float* W1   = (const float*)d_in[13];
  const float* b1   = (const float*)d_in[14];
  const float* W2   = (const float*)d_in[15];
  const float* b2   = (const float*)d_in[16];
  const float* ln2g = (const float*)d_in[17];
  const float* ln2b = (const float*)d_in[18];
  const float* Wf   = (const float*)d_in[19];
  const float* bfin = (const float*)d_in[20];
  float* out = (float*)d_out;

  const int    Nrow = B_ * S_;            // 2048
  const size_t NSD  = (size_t)Nrow * D_;  // 1.57M floats
  float* ws  = (float*)d_ws;
  float* hs  = ws;             // hidden state
  float* qb  = ws + 1 * NSD;
  float* kb  = ws + 2 * NSD;
  float* vb  = ws + 3 * NSD;
  float* cx  = ws + 4 * NSD;   // attention context
  float* t0  = ws + 5 * NSD;   // attn-out / ff-out temp
  float* h1  = ws + 6 * NSD;   // post-LN1
  float* ffm = ws + 7 * NSD;   // Nrow x F GELU activations

  const dim3 blk(256);
  const dim3 gD(D_ / 128, Nrow / 64);     // (6, 32)
  const dim3 gF(F_ / 128, Nrow / 64);     // (24, 32)
  const dim3 gV(V_ / 128, Nrow / 64);     // (250, 32)
  const dim3 gAttn(S_ / 64, B_ * H_);     // (16, 24)

  embed_kernel<<<Nrow, blk, 0, stream>>>(x, tok, pos, hs);

  for (int i = 0; i < L_; ++i) {
    const size_t oDD = (size_t)i * D_ * D_;
    const size_t oDF = (size_t)i * D_ * F_;
    const size_t oFD = (size_t)i * F_ * D_;
    gemm_bf16_wmma<<<gD, blk, 0, stream>>>(hs, Wq + oDD, bq + i * D_, qb, Nrow, D_, D_, 0);
    gemm_bf16_wmma<<<gD, blk, 0, stream>>>(hs, Wk + oDD, bk + i * D_, kb, Nrow, D_, D_, 0);
    gemm_bf16_wmma<<<gD, blk, 0, stream>>>(hs, Wv + oDD, bv + i * D_, vb, Nrow, D_, D_, 0);
    flash_attn_kernel<<<gAttn, blk, 0, stream>>>(qb, kb, vb, cx);
    gemm_bf16_wmma<<<gD, blk, 0, stream>>>(cx, Wo + oDD, bo + i * D_, t0, Nrow, D_, D_, 0);
    add_layernorm_kernel<<<Nrow, blk, 0, stream>>>(t0, hs, ln1g + i * D_, ln1b + i * D_, h1);
    gemm_bf16_wmma<<<gF, blk, 0, stream>>>(h1, W1 + oDF, b1 + i * F_, ffm, Nrow, D_, F_, 1);
    gemm_bf16_wmma<<<gD, blk, 0, stream>>>(ffm, W2 + oFD, b2 + i * D_, t0, Nrow, F_, D_, 0);
    add_layernorm_kernel<<<Nrow, blk, 0, stream>>>(t0, h1, ln2g + i * D_, ln2b + i * D_, hs);
  }

  // logits straight into d_out, log_softmax in place, hset appended
  gemm_bf16_wmma<<<gV, blk, 0, stream>>>(hs, Wf, bfin, out, Nrow, D_, V_, 0);
  log_softmax_kernel<<<Nrow, blk, 0, stream>>>(out);
  copy4_kernel<<<(int)((NSD / 4 + 255) / 256), blk, 0, stream>>>(
      hs, out + (size_t)Nrow * V_, (int)(NSD / 4));
}
